// triangle_mul_script_module_23562190586482
// MI455X (gfx1250) — matrix operand order repaired and checked by reading, not compiled, not run on hardware
//
#include <hip/hip_runtime.h>
#include <hip/hip_bf16.h>

typedef __attribute__((ext_vector_type(16))) __bf16 v16bf;
typedef __attribute__((ext_vector_type(8)))  __bf16 v8bf;
typedef __attribute__((ext_vector_type(8)))  float  v8f;

#define NN 512
#define CZ 128
#define CH 128
#define LDS_STRIDE 136   // bf16 elems per LDS row: 128 + 8 pad (bank-conflict relief)

__device__ __forceinline__ v16bf cat8(v8bf lo, v8bf hi) {
  return __builtin_shufflevector(lo, hi, 0,1,2,3,4,5,6,7,8,9,10,11,12,13,14,15);
}

__device__ __forceinline__ float sigmoidf_(float x) {
  return 1.0f / (1.0f + __expf(-x));
}

// ---------------------------------------------------------------------------
// Kernel 0: convert 4 fp32 weight matrices [CH][CZ] -> bf16, packed contiguously
// ---------------------------------------------------------------------------
__global__ __launch_bounds__(256)
void convert_weights_kernel(const float* __restrict__ wag, const float* __restrict__ wap,
                            const float* __restrict__ wbg, const float* __restrict__ wbp,
                            __bf16* __restrict__ wout) {
  int idx = blockIdx.x * blockDim.x + threadIdx.x;   // 0 .. 4*CH*CZ-1
  if (idx >= 4 * CH * CZ) return;
  int which = idx >> 14;          // CH*CZ == 16384
  int off   = idx & 16383;
  const float* src = (which == 0) ? wag : (which == 1) ? wap : (which == 2) ? wbg : wbp;
  wout[idx] = (__bf16)src[off];
}

// ---------------------------------------------------------------------------
// Kernel 1: fused projections + sigmoid gating.
// Each wave: 32 consecutive pixels (two 16-row A-tiles) x all 128 channels.
// Every weight B-fragment is reused by BOTH A-tiles -> 4 WMMAs per 4 frag loads,
// and two independent WMMAs are in flight per load batch (latency hiding).
//   A operand: z tile [16 x 32] bf16 (K-contiguous per half-wave, same order as B),
//   B operand: W rows, K-contiguous (lane n = channel n, holds 16 contiguous K).
//   acc initialized to bias (all 8 C/D values in a lane share channel lane%16).
// Stores a,b as bf16 into workspace laid out [c][i][k] (k contiguous):
// each lane's 8 D-rows are 8 consecutive k -> one 16-B store.
// ---------------------------------------------------------------------------
__global__ __launch_bounds__(128)
void proj_gate_kernel(const float* __restrict__ z, const float* __restrict__ mask,
                      const __bf16* __restrict__ wbf,  // [4][CH][CZ] bf16
                      const float* __restrict__ bag, const float* __restrict__ bap,
                      const float* __restrict__ bbg, const float* __restrict__ bbp,
                      __bf16* __restrict__ a_ws, __bf16* __restrict__ b_ws) {
  const int lane = threadIdx.x & 31;
  const int wave = (blockIdx.x * blockDim.x + threadIdx.x) >> 5;
  const int p0    = wave * 32;        // 32 consecutive pixels (same tensor row i)
  const int i     = p0 >> 9;          // p0 / NN
  const int k0col = p0 & (NN - 1);    // column of first pixel within row i

  // ---- A fragments, contiguous-K 16x32 layout (same K order as B):
  //      lane<16 holds K = {0..15}+kb ; lane>=16 holds K = {16..31}+kb
  const int koff = (lane >> 4) * 16;
  v16bf afrag[2][4];
#pragma unroll
  for (int mt = 0; mt < 2; ++mt) {
    const float* zrow = z + (size_t)(p0 + mt * 16 + (lane & 15)) * CZ;
#pragma unroll
    for (int kc = 0; kc < 4; ++kc) {
      const float* pz = zrow + kc * 32;
      float4 f0 = *(const float4*)(pz + koff);
      float4 f1 = *(const float4*)(pz + koff + 4);
      float4 f2 = *(const float4*)(pz + koff + 8);
      float4 f3 = *(const float4*)(pz + koff + 12);
      v16bf a;
      a[0]=(__bf16)f0.x;  a[1]=(__bf16)f0.y;  a[2]=(__bf16)f0.z;  a[3]=(__bf16)f0.w;
      a[4]=(__bf16)f1.x;  a[5]=(__bf16)f1.y;  a[6]=(__bf16)f1.z;  a[7]=(__bf16)f1.w;
      a[8]=(__bf16)f2.x;  a[9]=(__bf16)f2.y;  a[10]=(__bf16)f2.z; a[11]=(__bf16)f2.w;
      a[12]=(__bf16)f3.x; a[13]=(__bf16)f3.y; a[14]=(__bf16)f3.z; a[15]=(__bf16)f3.w;
      afrag[mt][kc] = a;
    }
  }

  // mask values for this lane's D rows (m = r + 8*(lane/16), pixel p0 + mt*16 + m)
  float mv[2][8];
#pragma unroll
  for (int mt = 0; mt < 2; ++mt) {
    const float* mrow = mask + p0 + mt * 16 + (lane >> 4) * 8;
    float4 m0 = *(const float4*)mrow;
    float4 m1 = *(const float4*)(mrow + 4);
    mv[mt][0]=m0.x; mv[mt][1]=m0.y; mv[mt][2]=m0.z; mv[mt][3]=m0.w;
    mv[mt][4]=m1.x; mv[mt][5]=m1.y; mv[mt][6]=m1.z; mv[mt][7]=m1.w;
  }

  const __bf16* Wag = wbf;
  const __bf16* Wap = wbf + CH * CZ;
  const __bf16* Wbg = wbf + 2 * CH * CZ;
  const __bf16* Wbp = wbf + 3 * CH * CZ;

  const int chl = lane & 15;        // channel within 16-wide tile (= D column)
  const int wko = (lane >> 4) * 16; // B-operand K half per half-wave

  for (int nt = 0; nt < 8; ++nt) {
    const int chbase = nt * 16;
    size_t ws_off[2];
#pragma unroll
    for (int mt = 0; mt < 2; ++mt)
      ws_off[mt] = ((size_t)(chbase + chl) * NN + i) * NN
                   + k0col + mt * 16 + (lane >> 4) * 8;

    // ---------------- a = mask * sigmoid(Wag z + bag) * (Wap z + bap)
    {
      float bgv = bag[chbase + chl];
      float bpv = bap[chbase + chl];
      v8f accg[2] = {{bgv,bgv,bgv,bgv,bgv,bgv,bgv,bgv},
                     {bgv,bgv,bgv,bgv,bgv,bgv,bgv,bgv}};
      v8f accp[2] = {{bpv,bpv,bpv,bpv,bpv,bpv,bpv,bpv},
                     {bpv,bpv,bpv,bpv,bpv,bpv,bpv,bpv}};
      const __bf16* wgrow = Wag + (size_t)(chbase + chl) * CZ + wko;
      const __bf16* wprow = Wap + (size_t)(chbase + chl) * CZ + wko;
#pragma unroll
      for (int kc = 0; kc < 4; ++kc) {
        v16bf bg_f = cat8(*(const v8bf*)(wgrow + kc * 32),
                          *(const v8bf*)(wgrow + kc * 32 + 8));
        v16bf bp_f = cat8(*(const v8bf*)(wprow + kc * 32),
                          *(const v8bf*)(wprow + kc * 32 + 8));
#pragma unroll
        for (int mt = 0; mt < 2; ++mt) {
          accg[mt] = __builtin_amdgcn_wmma_f32_16x16x32_bf16(
              false, afrag[mt][kc], false, bg_f, (short)0, accg[mt], false, false);
          accp[mt] = __builtin_amdgcn_wmma_f32_16x16x32_bf16(
              false, afrag[mt][kc], false, bp_f, (short)0, accp[mt], false, false);
        }
      }
#pragma unroll
      for (int mt = 0; mt < 2; ++mt) {
        v8bf av;
#pragma unroll
        for (int r = 0; r < 8; ++r)
          av[r] = (__bf16)(mv[mt][r] * sigmoidf_(accg[mt][r]) * accp[mt][r]);
        *(v8bf*)(a_ws + ws_off[mt]) = av;
      }
    }

    // ---------------- b = mask * sigmoid(Wbg z + bbg) * (Wbp z + bbp)
    {
      float bgv = bbg[chbase + chl];
      float bpv = bbp[chbase + chl];
      v8f accg[2] = {{bgv,bgv,bgv,bgv,bgv,bgv,bgv,bgv},
                     {bgv,bgv,bgv,bgv,bgv,bgv,bgv,bgv}};
      v8f accp[2] = {{bpv,bpv,bpv,bpv,bpv,bpv,bpv,bpv},
                     {bpv,bpv,bpv,bpv,bpv,bpv,bpv,bpv}};
      const __bf16* wgrow = Wbg + (size_t)(chbase + chl) * CZ + wko;
      const __bf16* wprow = Wbp + (size_t)(chbase + chl) * CZ + wko;
#pragma unroll
      for (int kc = 0; kc < 4; ++kc) {
        v16bf bg_f = cat8(*(const v8bf*)(wgrow + kc * 32),
                          *(const v8bf*)(wgrow + kc * 32 + 8));
        v16bf bp_f = cat8(*(const v8bf*)(wprow + kc * 32),
                          *(const v8bf*)(wprow + kc * 32 + 8));
#pragma unroll
        for (int mt = 0; mt < 2; ++mt) {
          accg[mt] = __builtin_amdgcn_wmma_f32_16x16x32_bf16(
              false, afrag[mt][kc], false, bg_f, (short)0, accg[mt], false, false);
          accp[mt] = __builtin_amdgcn_wmma_f32_16x16x32_bf16(
              false, afrag[mt][kc], false, bp_f, (short)0, accp[mt], false, false);
        }
      }
#pragma unroll
      for (int mt = 0; mt < 2; ++mt) {
        v8bf bv;
#pragma unroll
        for (int r = 0; r < 8; ++r)
          bv[r] = (__bf16)(mv[mt][r] * sigmoidf_(accg[mt][r]) * accp[mt][r]);
        *(v8bf*)(b_ws + ws_off[mt]) = bv;
      }
    }
  }
}

// ---------------------------------------------------------------------------
// Kernel 2: out[c,i,j] = sum_k a[c][i][k] * b[c][j][k]   (per-channel A @ B^T)
// Block = (channel, 128x64 output tile), 8 waves arranged 4x2, each wave owns a
// 32x32 patch (2x2 WMMA tiles). Per K-chunk: 2 A-frags + 2 B-frags -> 4 WMMAs
// (1 fragment load per WMMA). K staged through LDS in 128-wide slabs.
// ---------------------------------------------------------------------------
__global__ __launch_bounds__(256)
void triangle_outer_kernel(const __bf16* __restrict__ a_ws,
                           const __bf16* __restrict__ b_ws,
                           float* __restrict__ out) {
  __shared__ __attribute__((aligned(16))) __bf16 As[128 * LDS_STRIDE];
  __shared__ __attribute__((aligned(16))) __bf16 Bs[64 * LDS_STRIDE];

  const int c   = blockIdx.x >> 5;        // 128 channels, 32 tiles each
  const int rem = blockIdx.x & 31;
  const int it0 = (rem >> 3) * 128;       // 4 M-tiles of 128
  const int jt0 = (rem & 7) * 64;         // 8 N-tiles of 64

  const int lane = threadIdx.x & 31;
  const int wv   = threadIdx.x >> 5;      // 0..7
  const int wm   = (wv >> 1) * 32;        // wave M offset within block tile
  const int wn   = (wv & 1) * 32;         // wave N offset within block tile

  v8f acc[2][2] = {{{0,0,0,0,0,0,0,0},{0,0,0,0,0,0,0,0}},
                   {{0,0,0,0,0,0,0,0},{0,0,0,0,0,0,0,0}}};

  const __bf16* aBase = a_ws + (size_t)c * NN * NN;
  const __bf16* bBase = b_ws + (size_t)c * NN * NN;

  const int aoff = (lane >> 4) * 16;   // A-operand K half per half-wave
  const int boff = (lane >> 4) * 16;   // B-operand K half per half-wave
  const int rowl = lane & 15;

  for (int ks = 0; ks < 4; ++ks) {
    __syncthreads();
    // cooperative slab load: A 128 rows, B 64 rows, 128 k (bf16) each
#pragma unroll
    for (int t = threadIdx.x; t < 128 * 16; t += 256) {
      int row = t >> 4;
      int kc8 = t & 15;   // 8-elem (16B) chunk within the 128-wide slab
      *(uint4*)(&As[row * LDS_STRIDE + kc8 * 8]) =
          *(const uint4*)(aBase + (size_t)(it0 + row) * NN + ks * 128 + kc8 * 8);
      if (ks < 3)
        __builtin_prefetch(aBase + (size_t)(it0 + row) * NN + (ks + 1) * 128 + kc8 * 8, 0, 0);
    }
#pragma unroll
    for (int t = threadIdx.x; t < 64 * 16; t += 256) {
      int row = t >> 4;
      int kc8 = t & 15;
      *(uint4*)(&Bs[row * LDS_STRIDE + kc8 * 8]) =
          *(const uint4*)(bBase + (size_t)(jt0 + row) * NN + ks * 128 + kc8 * 8);
      if (ks < 3)
        __builtin_prefetch(bBase + (size_t)(jt0 + row) * NN + (ks + 1) * 128 + kc8 * 8, 0, 0);
    }
    __syncthreads();

#pragma unroll
    for (int kc = 0; kc < 4; ++kc) {
      const int kb = kc * 32;
      v16bf afr[2], bfr[2];
#pragma unroll
      for (int mi = 0; mi < 2; ++mi) {
        const __bf16* pa = &As[(wm + mi * 16 + rowl) * LDS_STRIDE + kb];
        afr[mi] = cat8(*(const v8bf*)(pa + aoff), *(const v8bf*)(pa + aoff + 8));
      }
#pragma unroll
      for (int ni = 0; ni < 2; ++ni) {
        const __bf16* pb = &Bs[(wn + ni * 16 + rowl) * LDS_STRIDE + kb + boff];
        bfr[ni] = cat8(*(const v8bf*)(pb), *(const v8bf*)(pb + 8));
      }
#pragma unroll
      for (int mi = 0; mi < 2; ++mi)
#pragma unroll
        for (int ni = 0; ni < 2; ++ni)
          acc[mi][ni] = __builtin_amdgcn_wmma_f32_16x16x32_bf16(
              false, afr[mi], false, bfr[ni], (short)0, acc[mi][ni], false, false);
    }
  }

  // D layout: lane covers column n = lane%16, rows m = r + 8*(lane/16)
  const int m0 = (lane >> 4) * 8;
  const int nj = lane & 15;
#pragma unroll
  for (int mi = 0; mi < 2; ++mi) {
#pragma unroll
    for (int ni = 0; ni < 2; ++ni) {
      float* o = out + ((size_t)c * NN + it0 + wm + mi * 16 + m0) * NN
                     + jt0 + wn + ni * 16 + nj;
#pragma unroll
      for (int r = 0; r < 8; ++r)
        o[(size_t)r * NN] = acc[mi][ni][r];
    }
  }
}

// ---------------------------------------------------------------------------
extern "C" void kernel_launch(void* const* d_in, const int* in_sizes, int n_in,
                              void* d_out, int out_size, void* d_ws, size_t ws_size,
                              hipStream_t stream) {
  (void)in_sizes; (void)n_in; (void)out_size; (void)ws_size;

  const float* z    = (const float*)d_in[0];
  const float* mask = (const float*)d_in[1];
  const float* wag  = (const float*)d_in[2];
  const float* bag  = (const float*)d_in[3];
  const float* wap  = (const float*)d_in[4];
  const float* bap  = (const float*)d_in[5];
  const float* wbg  = (const float*)d_in[6];
  const float* bbg  = (const float*)d_in[7];
  const float* wbp  = (const float*)d_in[8];
  const float* bbp  = (const float*)d_in[9];
  // d_in[10] = chunk_size: mathematically irrelevant (channels independent)

  // workspace layout:
  //   [0,            131072)              : 4 bf16 weight matrices
  //   [131072,       131072+64MiB)        : a_ws  bf16 [CH][NN][NN]
  //   [131072+64MiB, 131072+128MiB)       : b_ws  bf16 [CH][NN][NN]
  __bf16* wbf  = (__bf16*)d_ws;
  __bf16* a_ws = (__bf16*)((char*)d_ws + 4 * CH * CZ * sizeof(__bf16));
  __bf16* b_ws = a_ws + (size_t)CH * NN * NN;

  convert_weights_kernel<<<dim3((4 * CH * CZ + 255) / 256), dim3(256), 0, stream>>>(
      wag, wap, wbg, wbp, wbf);

  // 512*512 pixels / 32 per wave = 8192 waves; 4 waves (128 thr) per block
  proj_gate_kernel<<<dim3(2048), dim3(128), 0, stream>>>(
      z, mask, wbf, bag, bap, bbg, bbp, a_ws, b_ws);

  // 128 channels * (4 M-tiles x 8 N-tiles) of 128x64 output tiles
  triangle_outer_kernel<<<dim3(128 * 32), dim3(256), 0, stream>>>(
      a_ws, b_ws, (float*)d_out);
}
